// DynamicGroupMambaLayer_36472862277806
// MI455X (gfx1250) — compile-verified, matrix operand order checked by reading, not run on hardware
//
#include <hip/hip_runtime.h>
#include <hip/hip_bf16.h>

typedef __attribute__((ext_vector_type(16))) _Float16 v16h;
typedef __attribute__((ext_vector_type(8)))  _Float16 v8h;
typedef __attribute__((ext_vector_type(8)))  float    v8f;

#define BB 8
#define CC 384
#define LL 4096
#define DG 192
#define RR 12

__device__ __forceinline__ float sigmoidf_(float x){ return 1.0f / (1.0f + __expf(-x)); }
__device__ __forceinline__ float softplusf_(float x){ return x > 20.0f ? x : log1pf(__expf(x)); }

// ---- WMMA fragment load (CDNA5 ISA 7.12.2 layout) ---------------------------
// 16-bit A 16x32: lane row m=lane&15; lanes<16 hold K {0..7,16..23}, lanes>=16
// hold K {8..15,24..31}. Each half-fragment is 8 contiguous halfs = one b128.
__device__ __forceinline__ v16h load_frag16(const _Float16* __restrict__ p) {
  v8h lo = *(const v8h*)p;
  v8h hi = *(const v8h*)(p + 16);
  v16h r;
#pragma unroll
  for (int i = 0; i < 8; ++i) { r[i] = lo[i]; r[i + 8] = hi[i]; }
  return r;
}

// ---- blocked WMMA core: one wave computes a 16x64 output strip --------------
// A fragment is loaded once per K-step and reused across 4 column tiles:
// 4x the wmma work per byte of A traffic vs a single-tile wave.
struct Acc4 { v8f a[4]; };

__device__ __forceinline__ Acc4 wmma_strip4(const _Float16* __restrict__ A, int lda, int row0,
                                            const _Float16* __restrict__ W, int ldw, int col0,
                                            int Kdim, int lane) {
  Acc4 acc;
#pragma unroll
  for (int j = 0; j < 4; ++j) acc.a[j] = (v8f){};
  const int m = row0 + (lane & 15);
  const int nl = lane & 15;
  const int s = (lane >> 4) << 3;
  const _Float16* pa = A + (size_t)m * lda + s;
  const _Float16* pw0 = W + (size_t)(col0 + nl) * ldw + s;
  for (int k0 = 0; k0 < Kdim; k0 += 32) {
    __builtin_prefetch((const void*)(pa + 64), 0, 1);    // global_prefetch_b8
    v16h af = load_frag16(pa);
#pragma unroll
    for (int j = 0; j < 4; ++j) {
      v16h wf = load_frag16(pw0 + (size_t)(16 * j) * ldw);
      acc.a[j] = __builtin_amdgcn_wmma_f32_16x16x32_f16(false, af, false, wf,
                                                        (short)0, acc.a[j], false, false);
    }
    pa += 32; pw0 += 32;
  }
  return acc;
}

// ---- 1) LayerNorm over C + per-batch pool accumulation ----------------------
__global__ void k_ln_pool(const float* __restrict__ x, const float* __restrict__ g,
                          const float* __restrict__ bta, float* __restrict__ xn,
                          float* __restrict__ pool) {
  const int tok = blockIdx.x;            // B*L tokens
  const int b = tok / LL;
  const int tid = threadIdx.x;           // 128
  __shared__ float s1[128], s2[128];
  const float* xp = x + (size_t)tok * CC;
  float v0 = xp[tid], v1 = xp[tid + 128], v2 = xp[tid + 256];
  s1[tid] = v0 + v1 + v2;
  s2[tid] = v0 * v0 + v1 * v1 + v2 * v2;
  __syncthreads();
  for (int st = 64; st > 0; st >>= 1) {
    if (tid < st) { s1[tid] += s1[tid + st]; s2[tid] += s2[tid + st]; }
    __syncthreads();
  }
  const float mean = s1[0] * (1.0f / CC);
  const float var  = s2[0] * (1.0f / CC) - mean * mean;
  const float rs = rsqrtf(var + 1e-5f);
  float* xno = xn + (size_t)tok * CC;
#pragma unroll
  for (int i = 0; i < 3; ++i) {
    int c = tid + 128 * i;
    float val = (xp[c] - mean) * rs * g[c] + bta[c];
    xno[c] = val;
    atomicAdd(&pool[b * CC + c], val);
  }
}

// ---- 2) scoring MLP + stable descending argsort via rank counting -----------
__global__ void k_scores_sort(const float* __restrict__ pool,
                              const float* __restrict__ rw1, const float* __restrict__ rb1,
                              const float* __restrict__ rw2, const float* __restrict__ rb2,
                              int* __restrict__ sort_idx, int* __restrict__ inv_idx) {
  const int b = blockIdx.x;
  const int t = threadIdx.x;             // 384
  __shared__ float ps[CC];
  __shared__ float hid[DG];
  __shared__ float sc[CC];
  ps[t] = pool[b * CC + t] * (1.0f / LL);
  __syncthreads();
  if (t < DG) {
    float a = rb1[t];
    const float* w = rw1 + t * CC;
    for (int c = 0; c < CC; ++c) a += ps[c] * w[c];
    hid[t] = fmaxf(a, 0.0f);
  }
  __syncthreads();
  {
    float a = rb2[t];
    const float* w = rw2 + t * DG;
    for (int c = 0; c < DG; ++c) a += hid[c] * w[c];
    sc[t] = sigmoidf_(a);
  }
  __syncthreads();
  const float st = sc[t];
  int rank = 0;
  for (int j = 0; j < CC; ++j) {
    float sj = sc[j];
    if (sj > st || (sj == st && j < t)) ++rank;
  }
  sort_idx[b * CC + rank] = t;
  inv_idx[b * CC + t] = rank;
}

// ---- 3) weight conversion to f16 --------------------------------------------
__global__ void k_cvt_weights(const float* __restrict__ inw, const float* __restrict__ outw,
                              const float* __restrict__ projw,
                              _Float16* __restrict__ inw16, _Float16* __restrict__ outw16,
                              _Float16* __restrict__ projw16) {
  int i = blockIdx.x * blockDim.x + threadIdx.x;
  if (i < 2 * (2 * DG) * DG) inw16[i] = (_Float16)inw[i];
  if (i < 2 * DG * DG)       outw16[i] = (_Float16)outw[i];
  if (i < CC * CC)           projw16[i] = (_Float16)projw[i];
}

// ---- 4) channel gather by sort_idx + f16 convert -----------------------------
__global__ void k_gather(const float* __restrict__ xn, const int* __restrict__ sidx,
                         _Float16* __restrict__ xs16) {
  size_t i = (size_t)blockIdx.x * blockDim.x + threadIdx.x;  // B*L*C
  int c = (int)(i % CC);
  size_t tok = i / CC;
  int b = (int)(tok / LL);
  xs16[i] = (_Float16)xn[tok * CC + sidx[b * CC + c]];
}

// ---- 5) in_proj GEMM: (L x 192) @ (384 x 192)^T, 16x64 strip per wave -------
__global__ void k_gemm_inproj(const _Float16* __restrict__ xs16, const _Float16* __restrict__ w16,
                              float* __restrict__ xc, float* __restrict__ z) {
  const int lane = threadIdx.x & 31;
  const int wv = threadIdx.x >> 5;
  const int b = blockIdx.z >> 1, g = blockIdx.z & 1;
  const int row0 = (blockIdx.x * 8 + wv) * 16;
  const int col0 = blockIdx.y * 64;
  const _Float16* A = xs16 + (size_t)b * LL * CC + g * DG;
  const _Float16* W = w16 + (size_t)g * (2 * DG) * DG;
  Acc4 acc = wmma_strip4(A, CC, row0, W, DG, col0, DG, lane);
  const int nl = lane & 15;
  const int mb = row0 + ((lane >> 4) << 3);
  const size_t b2g = blockIdx.z;
#pragma unroll
  for (int j = 0; j < 4; ++j) {
    const int n = col0 + 16 * j + nl;
#pragma unroll
    for (int v = 0; v < 8; ++v) {
      int m = mb + v;
      float val = acc.a[j][v];
      if (n < DG) xc[(b2g * DG + n) * LL + m] = val;           // NCHW for conv/scan
      else        z[(b2g * LL + m) * DG + (n - DG)] = val;     // token-major gate
    }
  }
}

// ---- 6) depthwise 3x3 conv + bias + SiLU; writes scan-order layouts ----------
__global__ void k_conv_silu(const float* __restrict__ xc, const float* __restrict__ cw,
                            const float* __restrict__ cb, float* __restrict__ xconv,
                            _Float16* __restrict__ xcv16) {
  const size_t i = (size_t)blockIdx.x * blockDim.x + threadIdx.x;  // (b2g*192+d)*4096 + hw
  const int hw = (int)(i & 4095);
  const int cd = (int)(i >> 12);
  const int d = cd % DG;
  const int b2g = cd / DG;
  const int g = b2g & 1;
  const int h = hw >> 6, w = hw & 63;
  const float* src = xc + (size_t)cd * LL;
  const float* k9 = cw + (size_t)(g * DG + d) * 9;
  float a = cb[g * DG + d];
#pragma unroll
  for (int ky = 0; ky < 3; ++ky) {
    int yy = h + ky - 1;
    if (yy < 0 || yy > 63) continue;
#pragma unroll
    for (int kx = 0; kx < 3; ++kx) {
      int xx = w + kx - 1;
      if (xx < 0 || xx > 63) continue;
      a += k9[ky * 3 + kx] * src[yy * 64 + xx];
    }
  }
  a = a * sigmoidf_(a);
  const int lp = g ? (w * 64 + h) : hw;                 // vertical group scans transposed
  xconv[(size_t)cd * LL + lp] = a;
  xcv16[((size_t)b2g * LL + lp) * DG + d] = (_Float16)a;
}

// ---- 7) x_proj: 14 outputs per (b,g,k,l), flip handled by index --------------
__global__ void k_xproj(const _Float16* __restrict__ xcv16, const float* __restrict__ xpw,
                        float* __restrict__ xdbl) {
  const int i = blockIdx.x * blockDim.x + threadIdx.x;  // B*2*2*L
  const int l = i & 4095;
  const int k = (i >> 12) & 1;
  const int g = (i >> 13) & 1;
  const int b = i >> 14;
  const int b2g = b * 2 + g;
  const int lk = k ? (LL - 1 - l) : l;
  const _Float16* xv = xcv16 + ((size_t)b2g * LL + lk) * DG;
  float outv[14];
#pragma unroll
  for (int d = 0; d < 14; ++d) outv[d] = 0.0f;
  const float* w = xpw + (size_t)(g * 2 + k) * 14 * DG;
  for (int c = 0; c < DG; ++c) {
    float xvc = (float)xv[c];
#pragma unroll
    for (int d = 0; d < 14; ++d) outv[d] += xvc * w[d * DG + c];
  }
  float* o = xdbl + ((size_t)(b2g * 2 + k) * 14) * LL + l;
#pragma unroll
  for (int d = 0; d < 14; ++d) o[d * LL] = outv[d];
}

// ---- 8) selective scan (S=1): h = exp(dt*A)h + dt*B*x; y = h*C + D*x ---------
__global__ void k_scan(const float* __restrict__ xdbl, const float* __restrict__ xconv,
                       const float* __restrict__ dtw, const float* __restrict__ dtb,
                       const float* __restrict__ alog, const float* __restrict__ Ds,
                       float* __restrict__ ybuf) {
  const int i = blockIdx.x * blockDim.x + threadIdx.x;  // B*2*2*192 = 6144
  const int d = i % DG;
  const int k = (i / DG) & 1;
  const int g = (i / (2 * DG)) & 1;
  const int b = i / (4 * DG);
  const int b2g = b * 2 + g;
  const int gk = g * 2 + k;
  float wreg[RR];
#pragma unroll
  for (int r = 0; r < RR; ++r) wreg[r] = dtw[((size_t)gk * DG + d) * RR + r];
  const float biasdt = dtb[gk * DG + d];
  const float Av = -__expf(alog[gk * DG + d]);
  const float Dv = Ds[gk * DG + d];
  const float* xd = xdbl + ((size_t)(b2g * 2 + k) * 14) * LL;
  const float* xrow = xconv + ((size_t)b2g * DG + d) * LL;
  float* yrow = ybuf + ((size_t)b2g * DG + d) * LL;
  float hst = 0.0f;
  for (int l = 0; l < LL; ++l) {
    float acc = biasdt;
#pragma unroll
    for (int r = 0; r < RR; ++r) acc += xd[r * LL + l] * wreg[r];
    float dt = softplusf_(acc);
    float Bv = xd[12 * LL + l];
    float Cv = xd[13 * LL + l];
    int lk = k ? (LL - 1 - l) : l;
    float xv = xrow[lk];
    hst = __expf(dt * Av) * hst + dt * Bv * xv;
    atomicAdd(&yrow[lk], hst * Cv + Dv * xv);   // both directions accumulate
  }
}

// ---- 9) merge: out-LN over Di, * silu(z), -> f16 GEMM operand ----------------
__global__ void k_merge_ln(const float* __restrict__ ybuf, const float* __restrict__ z,
                           const float* __restrict__ ong, const float* __restrict__ onb,
                           _Float16* __restrict__ ygate16) {
  const int blk = blockIdx.x;            // B*2*4096
  const int t = blk & 4095;
  const int b2g = blk >> 12;
  const int g = b2g & 1;
  const int d = threadIdx.x;             // 192
  const int lpos = g ? ((t & 63) * 64 + (t >> 6)) : t;
  __shared__ float s1[DG], s2[DG];
  float v = ybuf[((size_t)b2g * DG + d) * LL + lpos];
  s1[d] = v; s2[d] = v * v;
  __syncthreads();
  if (d < 64) { s1[d] += s1[d + 128]; s2[d] += s2[d + 128]; }
  __syncthreads();
  if (d < 64) { s1[d] += s1[d + 64]; s2[d] += s2[d + 64]; }
  __syncthreads();
  for (int st = 32; st > 0; st >>= 1) {
    if (d < st) { s1[d] += s1[d + st]; s2[d] += s2[d + st]; }
    __syncthreads();
  }
  float mean = s1[0] * (1.0f / DG);
  float var  = s2[0] * (1.0f / DG) - mean * mean;
  float y = (v - mean) * rsqrtf(var + 1e-5f) * ong[g * DG + d] + onb[g * DG + d];
  float zv = z[((size_t)b2g * LL + t) * DG + d];
  ygate16[((size_t)b2g * LL + t) * DG + d] = (_Float16)(y * zv * sigmoidf_(zv));
}

// ---- 10) out_proj GEMM: (L x 192) @ (192 x 192)^T -> ycat, 16x64 strip ------
__global__ void k_gemm_outproj(const _Float16* __restrict__ ygate16,
                               const _Float16* __restrict__ ow16,
                               float* __restrict__ ycat) {
  const int lane = threadIdx.x & 31;
  const int wv = threadIdx.x >> 5;
  const int b2g = blockIdx.z;
  const int b = b2g >> 1, g = b2g & 1;
  const int row0 = (blockIdx.x * 8 + wv) * 16;
  const int col0 = blockIdx.y * 64;
  const _Float16* A = ygate16 + (size_t)b2g * LL * DG;
  const _Float16* W = ow16 + (size_t)g * DG * DG;
  Acc4 acc = wmma_strip4(A, DG, row0, W, DG, col0, DG, lane);
  const int nl = lane & 15;
  const int mb = row0 + ((lane >> 4) << 3);
#pragma unroll
  for (int j = 0; j < 4; ++j) {
    const int n = col0 + 16 * j + nl;
#pragma unroll
    for (int v = 0; v < 8; ++v) {
      ycat[((size_t)b * LL + mb + v) * CC + g * DG + n] = acc.a[j][v];
    }
  }
}

// ---- 11) channel-interaction gate + inverse permutation, -> f16 --------------
__global__ void k_gate_fuse(const float* __restrict__ xn, const float* __restrict__ ycat,
                            const int* __restrict__ inv_idx,
                            const float* __restrict__ cw1, const float* __restrict__ cb1,
                            const float* __restrict__ cw2, const float* __restrict__ cb2,
                            _Float16* __restrict__ A16) {
  const int tok = blockIdx.x;            // B*L
  const int b = tok / LL;
  const int c = threadIdx.x;             // 384
  __shared__ float xs_s[CC];
  __shared__ float hid[24];
  xs_s[c] = xn[(size_t)tok * CC + c];
  __syncthreads();
  if (c < 24) {
    float a = cb1[c];
    const float* w = cw1 + c * CC;
    for (int j = 0; j < CC; ++j) a += xs_s[j] * w[j];
    hid[c] = fmaxf(a, 0.0f);
  }
  __syncthreads();
  float a = cb2[c];
  const float* w = cw2 + c * 24;
#pragma unroll
  for (int j = 0; j < 24; ++j) a += hid[j] * w[j];
  float gate = sigmoidf_(a);
  float yr = ycat[(size_t)tok * CC + inv_idx[b * CC + c]];
  A16[(size_t)tok * CC + c] = (_Float16)(yr * gate);
}

// ---- 12) final proj GEMM + bias + residual + skip, 16x64 strip ---------------
__global__ void k_gemm_final(const _Float16* __restrict__ A16, const _Float16* __restrict__ pw16,
                             const float* __restrict__ pb, const float* __restrict__ x0,
                             const float* __restrict__ skip, float* __restrict__ out) {
  const int lane = threadIdx.x & 31;
  const int wv = threadIdx.x >> 5;
  const int row0 = (blockIdx.x * 8 + wv) * 16;
  const int col0 = blockIdx.y * 64;
  Acc4 acc = wmma_strip4(A16, CC, row0, pw16, CC, col0, CC, lane);
  const int nl = lane & 15;
  const int mb = row0 + ((lane >> 4) << 3);
  const float ss = skip[0];
#pragma unroll
  for (int j = 0; j < 4; ++j) {
    const int n = col0 + 16 * j + nl;
    const float bias = pb[n];
#pragma unroll
    for (int v = 0; v < 8; ++v) {
      size_t idx = (size_t)(mb + v) * CC + n;
      out[idx] = x0[idx] + ss * (acc.a[j][v] + bias);
    }
  }
}

extern "C" void kernel_launch(void* const* d_in, const int* in_sizes, int n_in,
                              void* d_out, int out_size, void* d_ws, size_t ws_size,
                              hipStream_t stream) {
  (void)in_sizes; (void)n_in; (void)out_size; (void)ws_size;
  const float* x      = (const float*)d_in[0];
  const float* norm_g = (const float*)d_in[1];
  const float* norm_b = (const float*)d_in[2];
  const float* r_w1   = (const float*)d_in[3];
  const float* r_b1   = (const float*)d_in[4];
  const float* r_w2   = (const float*)d_in[5];
  const float* r_b2   = (const float*)d_in[6];
  const float* in_w   = (const float*)d_in[7];
  const float* conv_w = (const float*)d_in[8];
  const float* conv_b = (const float*)d_in[9];
  const float* xp_w   = (const float*)d_in[10];
  const float* dt_w   = (const float*)d_in[11];
  const float* dt_b   = (const float*)d_in[12];
  const float* A_log  = (const float*)d_in[13];
  const float* Ds     = (const float*)d_in[14];
  const float* on_g   = (const float*)d_in[15];
  const float* on_b   = (const float*)d_in[16];
  const float* op_w   = (const float*)d_in[17];
  const float* ci_w1  = (const float*)d_in[18];
  const float* ci_b1  = (const float*)d_in[19];
  const float* ci_w2  = (const float*)d_in[20];
  const float* ci_b2  = (const float*)d_in[21];
  const float* proj_w = (const float*)d_in[22];
  const float* proj_b = (const float*)d_in[23];
  const float* skip   = (const float*)d_in[24];

  char* ws = (char*)d_ws;
  size_t off = 0;
  auto take = [&](size_t bytes) -> char* {
    char* p = ws + off;
    off = (off + bytes + 255) & ~(size_t)255;
    return p;
  };
  const size_t NTOK = (size_t)BB * LL;
  float*     xn    = (float*)take(NTOK * CC * 4);
  float*     pool  = (float*)take((size_t)BB * CC * 4);
  int*       sidx  = (int*)  take((size_t)BB * CC * 4);
  int*       iidx  = (int*)  take((size_t)BB * CC * 4);
  _Float16*  xs16  = (_Float16*)take(NTOK * CC * 2);
  _Float16*  in16  = (_Float16*)take((size_t)2 * (2 * DG) * DG * 2);
  _Float16*  out16 = (_Float16*)take((size_t)2 * DG * DG * 2);
  _Float16*  pw16  = (_Float16*)take((size_t)CC * CC * 2);
  float*     xc    = (float*)take((size_t)BB * 2 * DG * LL * 4);
  float*     zb    = (float*)take((size_t)BB * 2 * LL * DG * 4);
  float*     xconv = (float*)take((size_t)BB * 2 * DG * LL * 4);
  _Float16*  xcv16 = (_Float16*)take((size_t)BB * 2 * LL * DG * 2);
  float*     xdbl  = (float*)take((size_t)BB * 2 * 2 * 14 * LL * 4);
  float*     ybuf  = (float*)take((size_t)BB * 2 * DG * LL * 4);
  _Float16*  yg16  = (_Float16*)take((size_t)BB * 2 * LL * DG * 2);
  float*     ycat  = (float*)take(NTOK * CC * 4);
  _Float16*  a16   = (_Float16*)take(NTOK * CC * 2);

  hipMemsetAsync(pool, 0, (size_t)BB * CC * 4, stream);
  hipMemsetAsync(ybuf, 0, (size_t)BB * 2 * DG * LL * 4, stream);

  k_ln_pool<<<dim3(BB * LL), dim3(128), 0, stream>>>(x, norm_g, norm_b, xn, pool);
  k_scores_sort<<<dim3(BB), dim3(CC), 0, stream>>>(pool, r_w1, r_b1, r_w2, r_b2, sidx, iidx);
  k_cvt_weights<<<dim3(576), dim3(256), 0, stream>>>(in_w, op_w, proj_w, in16, out16, pw16);
  k_gather<<<dim3((unsigned)(NTOK * CC / 256)), dim3(256), 0, stream>>>(xn, sidx, xs16);
  k_gemm_inproj<<<dim3(LL / 128, (2 * DG) / 64, BB * 2), dim3(256), 0, stream>>>(xs16, in16, xc, zb);
  k_conv_silu<<<dim3((unsigned)((size_t)BB * 2 * DG * LL / 256)), dim3(256), 0, stream>>>(
      xc, conv_w, conv_b, xconv, xcv16);
  k_xproj<<<dim3(BB * 2 * 2 * LL / 256), dim3(256), 0, stream>>>(xcv16, xp_w, xdbl);
  k_scan<<<dim3(BB * 2 * 2 * DG / 256), dim3(256), 0, stream>>>(xdbl, xconv, dt_w, dt_b,
                                                                A_log, Ds, ybuf);
  k_merge_ln<<<dim3(BB * 2 * LL), dim3(DG), 0, stream>>>(ybuf, zb, on_g, on_b, yg16);
  k_gemm_outproj<<<dim3(LL / 128, DG / 64, BB * 2), dim3(256), 0, stream>>>(yg16, out16, ycat);
  k_gate_fuse<<<dim3(BB * LL), dim3(CC), 0, stream>>>(xn, ycat, iidx, ci_w1, ci_b1,
                                                      ci_w2, ci_b2, a16);
  k_gemm_final<<<dim3((BB * LL) / 128, CC / 64), dim3(256), 0, stream>>>(
      a16, pw16, proj_b, x, skip, (float*)d_out);
}
